// SelfAttention_12206297055644
// MI455X (gfx1250) — matrix operand order repaired and checked by reading, not compiled, not run on hardware
//
#include <hip/hip_runtime.h>

// ---------------------------------------------------------------------------
// Self-attention, B=4 S=4096 D=256, fp32 in/out.  MI455X / gfx1250 (wave32).
//   Pass 1: q/k/v = x@W + b via v_wmma_f32_16x16x32_f16 (f16 operands, fp32
//           accum). Q,K row-major f16; V transposed ([D,S] per batch).
//   Pass 2: flash attention. 1 workgroup = 4 waves = 64 queries (one 16-row
//           M-tile per wave). All waves share 32-key K/V blocks staged in LDS
//           by the Tensor Data Mover (tensor_load_to_lds, double-buffered,
//           s_wait_tensorcnt pipeline), with TDM LDS-padding to kill bank
//           conflicts. Per-wave online softmax; no cross-wave merge needed.
// ---------------------------------------------------------------------------

typedef _Float16 v16h __attribute__((ext_vector_type(16)));
typedef _Float16 v8h  __attribute__((ext_vector_type(8)));
typedef float    v8f  __attribute__((ext_vector_type(8)));
typedef unsigned int u32x4 __attribute__((ext_vector_type(4)));
typedef int          i32x8 __attribute__((ext_vector_type(8)));
typedef int          i32x4 __attribute__((ext_vector_type(4)));

#define DD    256
#define SS    4096
#define NB    4
#define NROWS (NB * SS)       // 16384
#define QB    64              // queries per workgroup
#define KBLK  32              // keys per block
#define NBLKS (SS / KBLK)     // 128

// LDS row strides (halves), padded +8 halves (4 DWORDs) per row so the 16
// lanes of a WMMA tile load hit distinct bank groups.
#define QROW  (DD + 8)        // 264
#define KROW  (DD + 8)        // 264
#define VROW  (KBLK + 8)      // 40
#define PROW  (KBLK + 8)      // 40

#define SMEM_HALVES (QB*QROW + 2*KBLK*KROW + 2*DD*VROW + 4*16*PROW)
#define SMEM_BYTES  (SMEM_HALVES * 2)

#if defined(__has_builtin)
#if __has_builtin(__builtin_amdgcn_tensor_load_to_lds)
#define HAVE_TDM 1
#endif
#endif
#ifndef HAVE_TDM
#define HAVE_TDM 0
#endif

#if HAVE_TDM
#if __has_builtin(__builtin_amdgcn_s_wait_tensorcnt)
#define TDM_WAIT(n) __builtin_amdgcn_s_wait_tensorcnt((short)(n))
#else
#define TDM_WAIT(n) asm volatile("s_wait_tensorcnt " #n ::: "memory")
#endif
#else
#define TDM_WAIT(n)
#endif

static __device__ __forceinline__ v8f zero_v8f() {
    v8f r;
#pragma unroll
    for (int i = 0; i < 8; ++i) r[i] = 0.0f;
    return r;
}

static __device__ __forceinline__ v16h cat8(v8h lo, v8h hi) {
    return __builtin_shufflevector(lo, hi, 0,1,2,3,4,5,6,7,8,9,10,11,12,13,14,15);
}

// WMMA A-operand (16x32 f16): lane row = lane%16; halves 0..15 -> K=g*16+0..15
// (contiguous, g = lane/16). rowStride in halves.
static __device__ __forceinline__ v16h loadA_f16(const _Float16* base, int rowStride,
                                                 int lr, int g) {
    const v8h* p0 = (const v8h*)(base + lr * rowStride + g * 16);
    const v8h* p1 = (const v8h*)(base + lr * rowStride + g * 16 + 8);
    return cat8(*p0, *p1);
}

// WMMA B-operand (32x16 f16): lane col = lane%16; halves 0..15 -> K=g*16+0..15
// (contiguous). rowPtr points at (K=0) of this lane's column walking K.
static __device__ __forceinline__ v16h loadB_f16(const _Float16* rowPtr, int g) {
    const v8h* p0 = (const v8h*)(rowPtr + g * 16);
    const v8h* p1 = (const v8h*)(rowPtr + g * 16 + 8);
    return cat8(*p0, *p1);
}

static __device__ __forceinline__ v8f wmma_f16(v16h a, v16h b, v8f c) {
    return __builtin_amdgcn_wmma_f32_16x16x32_f16(false, a, false, b, (short)0, c,
                                                  false, false);
}

// ---------------------------------------------------------------------------
// TDM 2D tile load: gPtr -> ldsPtr, tile (tile0 x tile1) of 2-byte elements,
// global row stride stride0 (elements), LDS rows padded to paddedRow halves
// via the D# pad fields (padInt: 2^padInt * 8B interval; padAmt: padAmt+1
// DWORDs inserted). Descriptor packing per CDNA5 ISA ch.8.
// This toolchain exposes the 6-arg builtin:
//   (u32x4 group0, i32x8 group1, i32x4 group2, i32x4 group3, i32x8, i32 cpol)
// groups 2/3 (and the trailing i32x8) are zero for 2D tiles.
// ---------------------------------------------------------------------------
static __device__ __forceinline__ void tdm_load_2d(
    void* ldsPtr, const void* gPtr,
    unsigned tile0, unsigned tile1,
    unsigned stride0,
    unsigned padInt, unsigned padAmt, unsigned paddedRow)
{
#if HAVE_TDM
    (void)paddedRow;
    unsigned lds = (unsigned)(size_t)ldsPtr;
    unsigned long long ga = (unsigned long long)(size_t)gPtr;
    u32x4 g0;
    g0[0] = 1u;                                             // count=1 (valid)
    g0[1] = lds;                                            // lds_addr
    g0[2] = (unsigned)ga;                                   // global_addr lo
    g0[3] = (unsigned)((ga >> 32) & 0x01FFFFFFull) | (2u << 30); // hi | type=2
    i32x8 g1;
    g1[0] = (int)((1u << 16) | (1u << 20) |                 // data_size=2B, pad_en
                  (padInt << 22) | (padAmt << 25));
    g1[1] = (int)((tile0 & 0xFFFFu) << 16);                 // tensor_dim0 lo16
    g1[2] = (int)((tile0 >> 16) | ((tile1 & 0xFFFFu) << 16)); // d0 hi | d1 lo
    g1[3] = (int)((tile1 >> 16) | ((tile0 & 0xFFFFu) << 16)); // d1 hi | tile_dim0
    g1[4] = (int)(tile1 & 0xFFFFu);                         // tile_dim1, dim2=0
    g1[5] = (int)stride0;                                   // dim0 stride lo32
    g1[6] = 0;                                              // stride hi, stride1 lo
    g1[7] = 0;
    i32x4 gz4;
    gz4[0] = 0; gz4[1] = 0; gz4[2] = 0; gz4[3] = 0;         // 2D: groups 2/3 zero
    i32x8 gz8;
#pragma unroll
    for (int i = 0; i < 8; ++i) gz8[i] = 0;
    __builtin_amdgcn_tensor_load_to_lds(g0, g1, gz4, gz4, gz8, 0);
#else
    // Cooperative fallback: calling wave (32 lanes) copies synchronously into
    // the same padded LDS layout.
    (void)padInt; (void)padAmt;
    const int lane = threadIdx.x & 31;
    _Float16* dst = (_Float16*)ldsPtr;
    const _Float16* src = (const _Float16*)gPtr;
    for (unsigned i = lane; i < tile0 * tile1; i += 32) {
        unsigned y = i / tile0, x = i - y * tile0;
        dst[y * paddedRow + x] = src[(size_t)y * stride0 + x];
    }
#endif
}

// ---------------------------------------------------------------------------
// Pass 1: QKV projection. grid = (NROWS/16, DD/16, 3), block = 32 (1 wave).
// ---------------------------------------------------------------------------
__global__ __launch_bounds__(32) void qkv_proj_kernel(
    const float* __restrict__ x,
    const float* __restrict__ Wq, const float* __restrict__ bq,
    const float* __restrict__ Wk, const float* __restrict__ bk,
    const float* __restrict__ Wv, const float* __restrict__ bv,
    _Float16* __restrict__ Qh, _Float16* __restrict__ Kh,
    _Float16* __restrict__ Vt)
{
    const int lane = threadIdx.x & 31;
    const int lr = lane & 15, g = lane >> 4;
    const int rt = blockIdx.x, ct = blockIdx.y, which = blockIdx.z;

    const float* W    = (which == 0) ? Wq : (which == 1) ? Wk : Wv;
    const float* bias = (which == 0) ? bq : (which == 1) ? bk : bv;

    const int row0 = rt * 16;
    const int col  = ct * 16 + lr;

    v8f acc = zero_v8f();

#pragma unroll
    for (int kk = 0; kk < DD / 32; ++kk) {
        const float* xr = x + (size_t)(row0 + lr) * DD + kk * 32;
        v16h a;
#pragma unroll
        for (int i = 0; i < 8; ++i) {
            a[i]     = (_Float16)xr[g * 16 + i];
            a[8 + i] = (_Float16)xr[g * 16 + 8 + i];
        }
        v16h b;
#pragma unroll
        for (int h = 0; h < 16; ++h)
            b[h] = (_Float16)W[(size_t)(kk * 32 + g * 16 + h) * DD + col];
        acc = wmma_f16(a, b, acc);
    }

    const float bcol = bias[col];
    if (which == 2) {
#pragma unroll
        for (int r = 0; r < 8; ++r) {
            int row = row0 + r + 8 * g;
            int bch = row >> 12;
            int srw = row & (SS - 1);
            Vt[(size_t)bch * DD * SS + (size_t)col * SS + srw] =
                (_Float16)(acc[r] + bcol);
        }
    } else {
        _Float16* o = (which == 0) ? Qh : Kh;
#pragma unroll
        for (int r = 0; r < 8; ++r)
            o[(size_t)(row0 + r + 8 * g) * DD + col] = (_Float16)(acc[r] + bcol);
    }
}

// ---------------------------------------------------------------------------
// Pass 2: flash attention. grid = NROWS/QB, block = 128 (4 waves),
// dynamic LDS = SMEM_BYTES. Wave w owns queries [qt*64 + w*16, +16).
// ---------------------------------------------------------------------------
__global__ __launch_bounds__(128) void flash_attn_kernel(
    const _Float16* __restrict__ Qh, const _Float16* __restrict__ Kh,
    const _Float16* __restrict__ Vt, float* __restrict__ out)
{
    extern __shared__ _Float16 smem[];

    const int tid  = threadIdx.x;
    const int w    = tid >> 5;
    const int lane = tid & 31;
    const int lr   = lane & 15, g = lane >> 4;
    const int qt   = blockIdx.x;
    const int bch  = (qt * QB) >> 12;

    _Float16* Qs  = smem;                                   // QB x QROW
    _Float16* Ks0 = Qs  + QB * QROW;                        // KBLK x KROW
    _Float16* Ks1 = Ks0 + KBLK * KROW;
    _Float16* Vs0 = Ks1 + KBLK * KROW;                      // DD x VROW
    _Float16* Vs1 = Vs0 + DD * VROW;
    _Float16* Pw  = Vs1 + DD * VROW + w * (16 * PROW);      // per-wave P

    const _Float16* Kb = Kh + (size_t)bch * SS * DD;
    const _Float16* Vb = Vt + (size_t)bch * DD * SS;
    const bool issuer = (tid < 32);

    // prologue: stage Q once, prime blocks 0 and 1 (double buffer)
    if (issuer) {
        tdm_load_2d(Qs,  Qh + (size_t)qt * QB * DD, DD, QB,   DD, 6, 3, QROW);
        tdm_load_2d(Ks0, Kb,                        DD, KBLK, DD, 6, 3, KROW);
        tdm_load_2d(Vs0, Vb,                        KBLK, DD, SS, 3, 3, VROW);
        tdm_load_2d(Ks1, Kb + (size_t)KBLK * DD,    DD, KBLK, DD, 6, 3, KROW);
        tdm_load_2d(Vs1, Vb + KBLK,                 KBLK, DD, SS, 3, 3, VROW);
    }

    v8f acc[16];
#pragma unroll
    for (int t = 0; t < 16; ++t) acc[t] = zero_v8f();
    float m_i[8], l_i[8];
#pragma unroll
    for (int r = 0; r < 8; ++r) { m_i[r] = -1.0e30f; l_i[r] = 0.0f; }

    const _Float16* Qa = Qs + (size_t)(w * 16) * QROW;

    for (int blk = 0; blk < NBLKS; ++blk) {
        _Float16* Ksb = (blk & 1) ? Ks1 : Ks0;
        _Float16* Vsb = (blk & 1) ? Vs1 : Vs0;
        if (issuer) {
            if (blk < NBLKS - 1) TDM_WAIT(2);   // oldest in-flight pair done
            else                 TDM_WAIT(0);
        }
        __syncthreads();                        // block (blk) visible to all

        // ---- scores: S[16x32] = Q(16xD) @ K_blk^T, K-loop over D ----
        v8f s0 = zero_v8f(), s1 = zero_v8f();
#pragma unroll
        for (int kk = 0; kk < DD / 32; ++kk) {
            v16h qa  = loadA_f16(Qa + kk * 32, QROW, lr, g);
            v16h kt0 = loadB_f16(Ksb + (size_t)lr * KROW + kk * 32, g);
            v16h kt1 = loadB_f16(Ksb + (size_t)(16 + lr) * KROW + kk * 32, g);
            s0 = wmma_f16(qa, kt0, s0);
            s1 = wmma_f16(qa, kt1, s1);
        }

        // ---- online softmax over this 32-key block ----
        float bm[8];
#pragma unroll
        for (int r = 0; r < 8; ++r) {
            s0[r] *= 0.0625f;                   // 1/sqrt(256)
            s1[r] *= 0.0625f;
            bm[r] = fmaxf(s0[r], s1[r]);
        }
#pragma unroll
        for (int mk = 1; mk < 16; mk <<= 1)
#pragma unroll
            for (int r = 0; r < 8; ++r)
                bm[r] = fmaxf(bm[r], __shfl_xor(bm[r], mk, 32));

        float corr[8], rs[8];
#pragma unroll
        for (int r = 0; r < 8; ++r) {
            float mn = fmaxf(m_i[r], bm[r]);
            corr[r] = __expf(m_i[r] - mn);
            m_i[r] = mn;
            float p0 = __expf(s0[r] - mn);
            float p1 = __expf(s1[r] - mn);
            s0[r] = p0; s1[r] = p1;
            rs[r] = p0 + p1;
        }
#pragma unroll
        for (int mk = 1; mk < 16; mk <<= 1)
#pragma unroll
            for (int r = 0; r < 8; ++r)
                rs[r] += __shfl_xor(rs[r], mk, 32);
#pragma unroll
        for (int r = 0; r < 8; ++r)
            l_i[r] = l_i[r] * corr[r] + rs[r];

#pragma unroll
        for (int t = 0; t < 16; ++t)
#pragma unroll
            for (int r = 0; r < 8; ++r)
                acc[t][r] *= corr[r];

        // ---- C-layout -> A-layout transpose of P via per-wave LDS ----
#pragma unroll
        for (int r = 0; r < 8; ++r) {
            Pw[(8 * g + r) * PROW + lr]      = (_Float16)s0[r];
            Pw[(8 * g + r) * PROW + 16 + lr] = (_Float16)s1[r];
        }
        v16h pa = loadA_f16(Pw, PROW, lr, g);   // same-wave DS ops in-order

        // ---- O[16xD] += P(16x32) @ V_blk(32xD) ----
#pragma unroll
        for (int t = 0; t < 16; ++t) {
            v16h vb = loadB_f16(Vsb + (size_t)(t * 16 + lr) * VROW, g);
            acc[t] = wmma_f16(pa, vb, acc[t]);
        }

        __syncthreads();                        // everyone done with buffer
        if (issuer && blk + 2 < NBLKS) {
            int nb = blk + 2;
            tdm_load_2d(Ksb, Kb + (size_t)nb * KBLK * DD, DD, KBLK, DD, 6, 3, KROW);
            tdm_load_2d(Vsb, Vb + nb * KBLK,              KBLK, DD, SS, 3, 3, VROW);
        }
    }

    // ---- normalize, write fp32 output (each wave owns its 16 rows) ----
#pragma unroll
    for (int t = 0; t < 16; ++t)
#pragma unroll
        for (int r = 0; r < 8; ++r)
            out[(size_t)(qt * QB + w * 16 + 8 * g + r) * DD + t * 16 + lr] =
                acc[t][r] / l_i[r];
}

// ---------------------------------------------------------------------------
extern "C" void kernel_launch(void* const* d_in, const int* in_sizes, int n_in,
                              void* d_out, int out_size, void* d_ws, size_t ws_size,
                              hipStream_t stream) {
    const float* x  = (const float*)d_in[0];
    const float* Wq = (const float*)d_in[1];
    const float* bq = (const float*)d_in[2];
    const float* Wk = (const float*)d_in[3];
    const float* bk = (const float*)d_in[4];
    const float* Wv = (const float*)d_in[5];
    const float* bv = (const float*)d_in[6];
    float* out = (float*)d_out;

    _Float16* Qh = (_Float16*)d_ws;                       // 8 MB
    _Float16* Kh = Qh + (size_t)NROWS * DD;               // 8 MB
    _Float16* Vt = Kh + (size_t)NROWS * DD;               // 8 MB (transposed)

    qkv_proj_kernel<<<dim3(NROWS / 16, DD / 16, 3), 32, 0, stream>>>(
        x, Wq, bq, Wk, bk, Wv, bv, Qh, Kh, Vt);
    flash_attn_kernel<<<dim3(NROWS / QB), 128, SMEM_BYTES, stream>>>(
        Qh, Kh, Vt, out);
}
